// tract_DGCNN_cls_70609262346927
// MI455X (gfx1250) — matrix operand order repaired and checked by reading, not compiled, not run on hardware
//
#include <hip/hip_runtime.h>
#include <hip/hip_bf16.h>

// ---------------------------------------------------------------------------
// DGCNN classifier forward for MI455X (gfx1250).
// Heavy GEMMs run on v_wmma_f32_16x16x32_bf16 (wave32 matrix cores).
// ---------------------------------------------------------------------------

#define LRELU_NEG 0.2f
#define BN_EPS    1e-5f

typedef __bf16 v16bf __attribute__((ext_vector_type(16)));
typedef float  v8f   __attribute__((ext_vector_type(8)));

union Frag {
    v16bf v;
    unsigned int u[8];
};

// ---------------------------------------------------------------------------
// KNN: one block per point set (32 points), pairwise -dist^2 via wave shuffles,
// stable top-K selection (lowest index wins ties -> matches jax.lax.top_k).
// ---------------------------------------------------------------------------
__global__ void knn_k(const float* __restrict__ X, int nStride, int C,
                      int* __restrict__ idxOut, int P, int Kn)
{
    const int n = blockIdx.x;
    const int p = threadIdx.x;          // 32 threads == one wave
    const float* bx = X + (long)n * nStride;

    float pd[32];
#pragma unroll
    for (int q = 0; q < 32; ++q) pd[q] = 0.f;

    for (int c = 0; c < C; ++c) {
        float xp = bx[c * P + p];
        for (int q = 0; q < 32; ++q) {
            float d = xp - __shfl(xp, q, 32);
            pd[q] -= d * d;
        }
    }
    for (int kk = 0; kk < Kn; ++kk) {
        float best = -1e38f; int bi = 0;
        for (int q = 0; q < 32; ++q)
            if (pd[q] > best) { best = pd[q]; bi = q; }
        idxOut[((long)n * P + p) * Kn + kk] = bi;
        pd[bi] = -1e30f;                // remove from further consideration
    }
}

// ---------------------------------------------------------------------------
// Weight conversion f32 -> bf16 with K padding to a multiple of 32.
// ---------------------------------------------------------------------------
__global__ void wconv_k(const float* __restrict__ src, __bf16* __restrict__ dst,
                        int rows, int kd, int kdpad)
{
    int t = blockIdx.x * 256 + threadIdx.x;
    if (t < rows * kdpad) {
        int r = t / kdpad, k = t - r * kdpad;
        dst[t] = (k < kd) ? (__bf16)src[r * kd + k] : (__bf16)0.f;
    }
}

__global__ void zero_k(float* __restrict__ p, int n)
{
    int i = blockIdx.x * 256 + threadIdx.x;
    if (i < n) p[i] = 0.f;
}

// ---------------------------------------------------------------------------
// WMMA GEMM. 2 waves / block, each wave owns a 16-row tile; loops over
// 16-column tiles x K-chunks of 32 (bf16). A-source modes:
//   0: edge features gathered from X=(n,cin,P) + knn idx  (staged in LDS)
//   1: rows (n,p) gathered from XC=(n,KD,P)               (staged in LDS)
//   2: plain row-major bf16 Arow (R x KDpad)              (direct loads)
// Epilogue: optional bias, bf16 and/or f32 store, fused per-channel sum/sumsq.
// ---------------------------------------------------------------------------
__global__ __launch_bounds__(64)
void wmma_gemm(int mode, long R, int KDpad, int Cout, int cin,
               const float* __restrict__ X, int nStrideX, const int* __restrict__ idxp,
               const float* __restrict__ XC, int nStrideC,
               const __bf16* __restrict__ Arow,
               const __bf16* __restrict__ Wb, const float* __restrict__ bias,
               __bf16* __restrict__ Y, float* __restrict__ Yf,
               float* __restrict__ gsum, float* __restrict__ gsumsq,
               int P, int Kn)
{
    __shared__ __bf16 sA[2][16 * 512];     // 32 KB: per-wave A tiles (modes 0/1)
    __shared__ float  ssum[1024];
    __shared__ float  ssumsq[1024];

    const int  wave    = threadIdx.x >> 5;
    const int  lane    = threadIdx.x & 31;
    const int  hi      = lane >> 4;        // 0: lanes 0-15, 1: lanes 16-31
    const int  lcol    = lane & 15;
    const long rowbase = ((long)blockIdx.x * 2 + wave) * 16;

    for (int c = threadIdx.x; c < 1024; c += 64) { ssum[c] = 0.f; ssumsq[c] = 0.f; }

    if (mode != 2) {
        __bf16* a = sA[wave];
        for (int row = 0; row < 16; ++row) {
            long r = rowbase + row;
            if (mode == 0) {
                long n  = r / ((long)P * Kn);
                int rem = (int)(r - n * (long)P * Kn);
                int p   = rem / Kn;
                int kk  = rem - p * Kn;
                const float* bx = X + n * (long)nStrideX;
                int j = idxp[(n * P + p) * (long)Kn + kk];
                for (int c = lane; c < KDpad; c += 32) {
                    float v = 0.f;
                    if (c < cin)          v = bx[c * P + j] - bx[c * P + p];
                    else if (c < 2 * cin) v = bx[(c - cin) * P + p];
                    a[row * KDpad + c] = (__bf16)v;
                }
            } else {
                long n = r / P;
                int  p = (int)(r - n * P);
                const float* bx = XC + n * (long)nStrideC;
                for (int c = lane; c < KDpad; c += 32)
                    a[row * KDpad + c] = (__bf16)bx[c * P + p];
            }
        }
    }
    __syncthreads();

    const int nkc = KDpad >> 5;
    const int nct = Cout >> 4;
    const unsigned int* arow32 =
        (mode == 2) ? (const unsigned int*)(Arow + (rowbase + lcol) * (long)KDpad)
                    : (const unsigned int*)(sA[wave] + lcol * KDpad);

    for (int ct = 0; ct < nct; ++ct) {
        v8f acc = {0.f, 0.f, 0.f, 0.f, 0.f, 0.f, 0.f, 0.f};
        const int ncol = ct * 16 + lcol;
        const unsigned int* wrow = (const unsigned int*)(Wb + (long)ncol * KDpad);

        for (int kc = 0; kc < nkc; ++kc) {
            Frag a, b;
#pragma unroll
            for (int v = 0; v < 8; ++v)             // B: column ncol, consecutive-K pairs
                b.u[v] = wrow[kc * 16 + hi * 8 + v];
#pragma unroll
            for (int v = 0; v < 8; ++v) {           // A: row lcol, consecutive-K pairs
                int dw = hi * 8 + v;
                a.u[v] = arow32[kc * 16 + dw];
            }
            acc = __builtin_amdgcn_wmma_f32_16x16x32_bf16(
                      false, a.v, false, b.v, (short)0, acc, false, false);
        }

        float bv = bias ? bias[ncol] : 0.f;
        float s = 0.f, sq = 0.f;
#pragma unroll
        for (int v = 0; v < 8; ++v) {               // D: VGPR v -> row v (+8 for hi lanes)
            float val = acc[v] + bv;
            long  rr  = rowbase + v + hi * 8;
            long  o   = rr * Cout + ncol;
            if (Y)  Y[o]  = (__bf16)val;
            if (Yf) Yf[o] = val;
            s += val; sq += val * val;
        }
        if (gsum) {
            s  += __shfl_xor(s,  16, 32);
            sq += __shfl_xor(sq, 16, 32);
            if (hi == 0) { atomicAdd(&ssum[ncol], s); atomicAdd(&ssumsq[ncol], sq); }
        }
    }
    __syncthreads();
    if (gsum) {
        for (int c = threadIdx.x; c < Cout; c += 64) {
            atomicAdd(&gsum[c],   ssum[c]);
            atomicAdd(&gsumsq[c], ssumsq[c]);
        }
    }
}

// ---------------------------------------------------------------------------
// Fold channel sums into BN scale/shift: y_bn = y*scale + shift.
// ---------------------------------------------------------------------------
__global__ void stats_k(const float* __restrict__ sum, const float* __restrict__ sumsq,
                        const float* __restrict__ g, const float* __restrict__ b,
                        float* __restrict__ scale, float* __restrict__ shift,
                        int C, float invR)
{
    int c = blockIdx.x * blockDim.x + threadIdx.x;
    if (c < C) {
        float m  = sum[c] * invR;
        float v  = sumsq[c] * invR - m * m;
        float sc = g[c] * rsqrtf(v + BN_EPS);
        scale[c] = sc;
        shift[c] = b[c] - m * sc;
    }
}

// BN + leaky-relu + max over K neighbors; writes (n, c, p) into concat buffer.
__global__ void bn_max_k(const __bf16* __restrict__ Y,
                         const float* __restrict__ scale, const float* __restrict__ shift,
                         float* __restrict__ Xout, long total,
                         int Cout, int P, int Kn, int nStrideOut)
{
    long t = (long)blockIdx.x * 256 + threadIdx.x;
    if (t < total) {
        long n  = t / ((long)Cout * P);
        int  cp = (int)(t - n * (long)Cout * P);
        int  c  = cp / P, p = cp - (cp / P) * P;
        float sc = scale[c], sh = shift[c];
        const __bf16* bp = Y + ((n * P + p) * (long)Kn) * Cout + c;
        float mx = -3.4e38f;
        for (int kk = 0; kk < Kn; ++kk) {
            float v = (float)bp[(long)kk * Cout] * sc + sh;
            v = v > 0.f ? v : LRELU_NEG * v;
            mx = fmaxf(mx, v);
        }
        Xout[n * (long)nStrideOut + c * P + p] = mx;
    }
}

// BN + leaky-relu on conv5 output, then max & mean pool over P -> h (bf16).
__global__ void bn5_pool_k(const __bf16* __restrict__ Y,
                           const float* __restrict__ scale, const float* __restrict__ shift,
                           __bf16* __restrict__ hb, int N, int C, int P)
{
    int t = blockIdx.x * 256 + threadIdx.x;
    if (t < N * C) {
        int n = t / C, c = t - (t / C) * C;
        float sc = scale[c], sh = shift[c];
        const __bf16* bp = Y + (long)n * P * C + c;
        float mx = -3.4e38f, sum = 0.f;
        for (int p = 0; p < P; ++p) {
            float v = (float)bp[(long)p * C] * sc + sh;
            v = v > 0.f ? v : LRELU_NEG * v;
            mx = fmaxf(mx, v);
            sum += v;
        }
        hb[(long)n * 2 * C + c]     = (__bf16)mx;
        hb[(long)n * 2 * C + C + c] = (__bf16)(sum / (float)P);
    }
}

// Elementwise BN + leaky-relu -> bf16 (operand of next FC).
__global__ void bn_act_k(const __bf16* __restrict__ Y,
                         const float* __restrict__ scale, const float* __restrict__ shift,
                         __bf16* __restrict__ out, int total, int C)
{
    int t = blockIdx.x * 256 + threadIdx.x;
    if (t < total) {
        int c = t % C;
        float v = (float)Y[t] * scale[c] + shift[c];
        v = v > 0.f ? v : LRELU_NEG * v;
        out[t] = (__bf16)v;
    }
}

// Row-wise log_softmax over C columns.
__global__ void logsoftmax_k(const float* __restrict__ logits,
                             float* __restrict__ out, int C)
{
    __shared__ float red[128];
    const int row = blockIdx.x, tid = threadIdx.x;
    const float* lp = logits + (long)row * C;

    float mx = -3.4e38f;
    for (int c = tid; c < C; c += 128) mx = fmaxf(mx, lp[c]);
    red[tid] = mx; __syncthreads();
    for (int s = 64; s > 0; s >>= 1) {
        if (tid < s) red[tid] = fmaxf(red[tid], red[tid + s]);
        __syncthreads();
    }
    mx = red[0]; __syncthreads();

    float sum = 0.f;
    for (int c = tid; c < C; c += 128) sum += expf(lp[c] - mx);
    red[tid] = sum; __syncthreads();
    for (int s = 64; s > 0; s >>= 1) {
        if (tid < s) red[tid] += red[tid + s];
        __syncthreads();
    }
    float ls = logf(red[0]);
    for (int c = tid; c < C; c += 128)
        out[(long)row * C + c] = lp[c] - mx - ls;
}

// ---------------------------------------------------------------------------
// Host orchestration.
// ---------------------------------------------------------------------------
extern "C" void kernel_launch(void* const* d_in, const int* in_sizes, int n_in,
                              void* d_out, int out_size, void* d_ws, size_t ws_size,
                              hipStream_t stream)
{
    (void)in_sizes; (void)n_in; (void)out_size; (void)ws_size;

    const float* x   = (const float*)d_in[0];
    const float* W1  = (const float*)d_in[2];
    const float* g1  = (const float*)d_in[3];
    const float* b1  = (const float*)d_in[4];
    const float* W2  = (const float*)d_in[5];
    const float* g2  = (const float*)d_in[6];
    const float* b2  = (const float*)d_in[7];
    const float* W3  = (const float*)d_in[8];
    const float* g3  = (const float*)d_in[9];
    const float* b3  = (const float*)d_in[10];
    const float* W4  = (const float*)d_in[11];
    const float* g4  = (const float*)d_in[12];
    const float* b4  = (const float*)d_in[13];
    const float* W5  = (const float*)d_in[14];
    const float* g5  = (const float*)d_in[15];
    const float* b5  = (const float*)d_in[16];
    const float* Wl1 = (const float*)d_in[17];
    const float* g6  = (const float*)d_in[18];
    const float* b6  = (const float*)d_in[19];
    const float* Wl2 = (const float*)d_in[20];
    const float* bl2 = (const float*)d_in[21];
    const float* g7  = (const float*)d_in[22];
    const float* b7  = (const float*)d_in[23];
    const float* Wl3 = (const float*)d_in[24];
    const float* bl3 = (const float*)d_in[25];

    const int  N = 2048, P = 32, Kn = 15;
    const long RE = (long)N * P * Kn;   // 983040 edge rows
    const long R5 = (long)N * P;        // 65536 point rows

    char* ws = (char*)d_ws;
    size_t off = 0;
    auto alloc = [&](size_t bytes) -> void* {
        void* p = ws + off;
        off += (bytes + 255) & ~(size_t)255;
        return p;
    };

    int*    idxb = (int*)   alloc((size_t)RE * sizeof(int));
    __bf16* Wb1  = (__bf16*)alloc((size_t)64   * 32   * 2);
    __bf16* Wb2  = (__bf16*)alloc((size_t)64   * 128  * 2);
    __bf16* Wb3  = (__bf16*)alloc((size_t)128  * 128  * 2);
    __bf16* Wb4  = (__bf16*)alloc((size_t)256  * 256  * 2);
    __bf16* Wb5  = (__bf16*)alloc((size_t)1024 * 512  * 2);
    __bf16* WbL1 = (__bf16*)alloc((size_t)512  * 2048 * 2);
    __bf16* WbL2 = (__bf16*)alloc((size_t)256  * 512  * 2);
    __bf16* WbL3 = (__bf16*)alloc((size_t)800  * 256  * 2);
    __bf16* Ybuf = (__bf16*)alloc((size_t)RE * 256 * 2);          // largest GEMM output
    float*  xc   = (float*) alloc((size_t)N * 512 * P * 4);       // concat [x1..x4]
    __bf16* hb   = (__bf16*)alloc((size_t)N * 2048 * 2);
    __bf16* h1b  = (__bf16*)alloc((size_t)N * 512 * 2);
    __bf16* h2b  = (__bf16*)alloc((size_t)N * 256 * 2);
    float*  logits = (float*)alloc((size_t)N * 800 * 4);
    float*  statb  = (float*)alloc((size_t)4096 * 4);
    float* gsum = statb, *gsumsq = statb + 1024, *scal = statb + 2048, *shif = statb + 3072;

    // --- weights -> bf16 (K padded to multiple of 32) ---
    auto cvt = [&](const float* src, __bf16* dst, int rows, int kd, int kdpad) {
        int tot = rows * kdpad;
        wconv_k<<<(tot + 255) / 256, 256, 0, stream>>>(src, dst, rows, kd, kdpad);
    };
    cvt(W1,  Wb1,  64,   6,    32);
    cvt(W2,  Wb2,  64,   128,  128);
    cvt(W3,  Wb3,  128,  128,  128);
    cvt(W4,  Wb4,  256,  256,  256);
    cvt(W5,  Wb5,  1024, 512,  512);
    cvt(Wl1, WbL1, 512,  2048, 2048);
    cvt(Wl2, WbL2, 256,  512,  512);
    cvt(Wl3, WbL3, 800,  256,  256);

    // --- edge blocks ---
    auto edge_block = [&](const float* Xin, int nStrideX, int cin, const __bf16* Wb,
                          int Cout, const float* g, const float* bta, float* Xout) {
        knn_k<<<N, 32, 0, stream>>>(Xin, nStrideX, cin, idxb, P, Kn);
        zero_k<<<8, 256, 0, stream>>>(gsum, 2048);
        int kdpad = 2 * cin; if (kdpad < 32) kdpad = 32;
        wmma_gemm<<<(int)(RE / 32), 64, 0, stream>>>(0, RE, kdpad, Cout, cin,
            Xin, nStrideX, idxb, nullptr, 0, nullptr, Wb, nullptr,
            Ybuf, nullptr, gsum, gsumsq, P, Kn);
        stats_k<<<(Cout + 63) / 64, 64, 0, stream>>>(gsum, gsumsq, g, bta,
                                                     scal, shif, Cout, 1.f / (float)RE);
        long tot = (long)N * Cout * P;
        bn_max_k<<<(int)((tot + 255) / 256), 256, 0, stream>>>(Ybuf, scal, shif,
            Xout, tot, Cout, P, Kn, 512 * P);
    };
    edge_block(x,            3 * P,   3,   Wb1, 64,  g1, b1, xc + 0);
    edge_block(xc + 0,       512 * P, 64,  Wb2, 64,  g2, b2, xc + 64  * P);
    edge_block(xc + 64 * P,  512 * P, 64,  Wb3, 128, g3, b3, xc + 128 * P);
    edge_block(xc + 128 * P, 512 * P, 128, Wb4, 256, g4, b4, xc + 256 * P);

    // --- conv5: (n,p) x 512 -> 1024, BN + lrelu + max/mean pool over P ---
    zero_k<<<8, 256, 0, stream>>>(gsum, 2048);
    wmma_gemm<<<(int)(R5 / 32), 64, 0, stream>>>(1, R5, 512, 1024, 0,
        nullptr, 0, nullptr, xc, 512 * P, nullptr, Wb5, nullptr,
        Ybuf, nullptr, gsum, gsumsq, P, Kn);
    stats_k<<<16, 64, 0, stream>>>(gsum, gsumsq, g5, b5, scal, shif, 1024, 1.f / (float)R5);
    bn5_pool_k<<<(N * 1024 + 255) / 256, 256, 0, stream>>>(Ybuf, scal, shif, hb, N, 1024, P);

    // --- FC head ---
    auto fc = [&](const __bf16* Ain, int kd, const __bf16* Wb, int Cout,
                  const float* bias, __bf16* Yo, float* Yfo, bool bn) {
        if (bn) zero_k<<<8, 256, 0, stream>>>(gsum, 2048);
        wmma_gemm<<<N / 32, 64, 0, stream>>>(2, (long)N, kd, Cout, 0,
            nullptr, 0, nullptr, nullptr, 0, Ain, Wb, bias, Yo, Yfo,
            bn ? gsum : nullptr, bn ? gsumsq : nullptr, P, Kn);
    };
    fc(hb, 2048, WbL1, 512, nullptr, Ybuf, nullptr, true);
    stats_k<<<8, 64, 0, stream>>>(gsum, gsumsq, g6, b6, scal, shif, 512, 1.f / (float)N);
    bn_act_k<<<(N * 512 + 255) / 256, 256, 0, stream>>>(Ybuf, scal, shif, h1b, N * 512, 512);

    fc(h1b, 512, WbL2, 256, bl2, Ybuf, nullptr, true);
    stats_k<<<4, 64, 0, stream>>>(gsum, gsumsq, g7, b7, scal, shif, 256, 1.f / (float)N);
    bn_act_k<<<(N * 256 + 255) / 256, 256, 0, stream>>>(Ybuf, scal, shif, h2b, N * 256, 256);

    fc(h2b, 256, WbL3, 800, bl3, nullptr, logits, false);

    logsoftmax_k<<<N, 128, 0, stream>>>(logits, (float*)d_out, 800);
}
